// AxialAttention_89069031784985
// MI455X (gfx1250) — matrix operand order repaired and checked by reading, not compiled, not run on hardware
//
#include <hip/hip_runtime.h>
#include <hip/hip_bf16.h>

typedef __attribute__((ext_vector_type(16))) __bf16 v16bf;
typedef __attribute__((ext_vector_type(8)))  float  v8f;
typedef __attribute__((ext_vector_type(8)))  unsigned short v8us;

union ABf { v16bf v; v8us h[2]; unsigned short u[16]; };

__device__ __forceinline__ unsigned short f2bf(float f) {
  unsigned int x = __float_as_uint(f);
  x += 0x7FFFu + ((x >> 16) & 1u);   // round-to-nearest-even
  return (unsigned short)(x >> 16);
}

#define WMMA_BF16(A, B, C) \
  __builtin_amdgcn_wmma_f32_16x16x32_bf16(false, (A), false, (B), (short)0, (C), false, false)

// sched_group_barrier masks: 0x008 = MFMA/WMMA, 0x020 = VMEM read, 0x100 = DS read
#if defined(__has_builtin)
#if __has_builtin(__builtin_amdgcn_sched_group_barrier)
#define SGB(mask, cnt, id) __builtin_amdgcn_sched_group_barrier((mask), (cnt), (id))
#endif
#endif
#ifndef SGB
#define SGB(mask, cnt, id)
#endif

// ---------------- weight prep: fp32 -> bf16, transposed ----------------
// ws layout (ushort): [qkvT pass0: 768x256][qkvT pass1: 768x256][outT pass0: 256x256][outT pass1: 256x256]
__global__ void prep_weights_kernel(const float* __restrict__ Wq0, const float* __restrict__ Wkv0,
                                    const float* __restrict__ Wout0,
                                    const float* __restrict__ Wq1, const float* __restrict__ Wkv1,
                                    const float* __restrict__ Wout1,
                                    unsigned short* __restrict__ ws) {
  const int QKV = 768 * 256, OUT = 256 * 256;
  int idx = blockIdx.x * blockDim.x + threadIdx.x;
  if (idx < QKV) {
    int o = idx >> 8, k = idx & 255;
    float v = (o < 256) ? Wq0[k * 256 + o] : Wkv0[k * 512 + (o - 256)];
    ws[idx] = f2bf(v);
  } else if (idx < 2 * QKV) {
    int j = idx - QKV, o = j >> 8, k = j & 255;
    float v = (o < 256) ? Wq1[k * 256 + o] : Wkv1[k * 512 + (o - 256)];
    ws[idx] = f2bf(v);
  } else if (idx < 2 * QKV + OUT) {
    int j = idx - 2 * QKV, o = j >> 8, k = j & 255;
    ws[idx] = f2bf(Wout0[k * 256 + o]);
  } else if (idx < 2 * QKV + 2 * OUT) {
    int j = idx - 2 * QKV - OUT, o = j >> 8, k = j & 255;
    ws[idx] = f2bf(Wout1[k * 256 + o]);
  }
}

// ---------------- fused axial attention, one block = one sequence ----------------
// T=128 tokens, C=256, HEADS=8, E=32. 256 threads = 8 waves; wave w owns token rows 16w..16w+15.
__global__ __launch_bounds__(256)
void axial_attn_kernel(const float* __restrict__ x,
                       const unsigned short* __restrict__ wqkvT,   // [768][256] bf16
                       const unsigned short* __restrict__ woutT,   // [256][256] bf16
                       const float* __restrict__ bout,             // [256] fp32
                       float* __restrict__ out,
                       int mode)                                   // 0 = H-pass (write), 1 = W-pass (accumulate)
{
  __shared__ __align__(16) unsigned short sK[128 * 264];   // K row-major [tok][c], pad 8
  __shared__ __align__(16) unsigned short sV[256 * 136];   // V transposed [c][tok], pad 8
  __shared__ __align__(16) unsigned short sScr[8 * 512];   // per-wave 16x32 transpose scratch

  const int wv   = threadIdx.x >> 5;
  const int lane = threadIdx.x & 31;
  const int l16  = lane & 15;
  const int half = lane >> 4;
  const int r0   = wv * 16;

  size_t base; int tokStride;
  if (mode == 0) { // seq = (b,w), tokens along H: x[b][j][w][c]
    int b = blockIdx.x >> 7, w = blockIdx.x & 127;
    base = (size_t)b * (128 * 128 * 256) + (size_t)w * 256;
    tokStride = 128 * 256;
  } else {         // seq = (b,h), tokens along W: x[b][h][j][c]
    base = (size_t)blockIdx.x * (128 * 256);
    tokStride = 256;
  }

  // ---- Phase A: load this wave's 16 X rows into registers (A-layout, bf16) ----
  ABf xa[8];
  const float* xr = x + base + (size_t)(r0 + l16) * tokStride;
  #pragma unroll
  for (int kk = 0; kk < 8; ++kk) {
    const float* p0 = xr + kk * 32 + half * 8;       // K = kk*32 + half*8 + {0..7}
    const float* p1 = xr + kk * 32 + 16 + half * 8;  // K = kk*32 + 16 + half*8 + {0..7}
    float4 f0 = *(const float4*)(p0);
    float4 f1 = *(const float4*)(p0 + 4);
    float4 f2 = *(const float4*)(p1);
    float4 f3 = *(const float4*)(p1 + 4);
    xa[kk].u[0]  = f2bf(f0.x); xa[kk].u[1]  = f2bf(f0.y); xa[kk].u[2]  = f2bf(f0.z); xa[kk].u[3]  = f2bf(f0.w);
    xa[kk].u[4]  = f2bf(f1.x); xa[kk].u[5]  = f2bf(f1.y); xa[kk].u[6]  = f2bf(f1.z); xa[kk].u[7]  = f2bf(f1.w);
    xa[kk].u[8]  = f2bf(f2.x); xa[kk].u[9]  = f2bf(f2.y); xa[kk].u[10] = f2bf(f2.z); xa[kk].u[11] = f2bf(f2.w);
    xa[kk].u[12] = f2bf(f3.x); xa[kk].u[13] = f2bf(f3.y); xa[kk].u[14] = f2bf(f3.z); xa[kk].u[15] = f2bf(f3.w);
  }

  // ---- Phase B: QKV = X @ [Wq|Wk|Wv]^T ----
  // Double-buffered B fragments + sched_group_barrier so each tile is a clause of
  // 16 global_load_b128 followed by 8 back-to-back WMMAs; loads for tile n+1 fly
  // while WMMAs of tile n execute.
  ABf qa[8];                       // Q rows kept in registers, A-layout, one v16bf per head
  unsigned short* scr = sScr + wv * 512;
  const unsigned short* wbase = wqkvT + (size_t)l16 * 256 + half * 8;

  auto loadB = [&](ABf (&b)[8], int nt) {
    const unsigned short* wrow = wbase + (size_t)nt * (16 * 256);
    #pragma unroll
    for (int kk = 0; kk < 8; ++kk) {
      b[kk].h[0] = *(const v8us*)(wrow + kk * 32);
      b[kk].h[1] = *(const v8us*)(wrow + kk * 32 + 16);
    }
  };
  auto tileStore = [&](const v8f& acc, int nt) {
    if (nt < 16) {
      // Q tile: C-layout -> A-layout via per-wave LDS scratch transpose
      #pragma unroll
      for (int i = 0; i < 8; ++i)
        scr[(i + half * 8) * 32 + l16] = f2bf(acc[i]);
      asm volatile("s_wait_dscnt 0x0" ::: "memory");
      qa[nt >> 1].h[nt & 1] = *(const v8us*)(scr + l16 * 32 + half * 8);
    } else if (nt < 32) {
      // K tile: row-major [token][c]
      int c0 = (nt - 16) * 16 + l16;
      #pragma unroll
      for (int i = 0; i < 8; ++i)
        sK[(r0 + i + half * 8) * 264 + c0] = f2bf(acc[i]);
    } else {
      // V tile: transposed [c][token]
      int c0 = (nt - 32) * 16 + l16;
      #pragma unroll
      for (int i = 0; i < 8; ++i)
        sV[c0 * 136 + r0 + i + half * 8] = f2bf(acc[i]);
    }
  };

  ABf bA[8], bB[8];
  loadB(bA, 0);
  for (int nt = 0; nt < 48; nt += 2) {
    loadB(bB, nt + 1);
    {
      v8f acc = {};
      #pragma unroll
      for (int kk = 0; kk < 8; ++kk) acc = WMMA_BF16(xa[kk].v, bA[kk].v, acc);
      SGB(0x020, 16, 0);   // 16 VMEM reads (next buffer) first
      SGB(0x008, 8, 0);    // then 8 WMMAs on current buffer
      tileStore(acc, nt);
    }
    // prefetch one tile ahead; at nt==46 this reads 8KB past this pass's qkvT,
    // which stays inside the 1MB weight region of d_ws (harmless).
    loadB(bA, nt + 2);
    {
      v8f acc = {};
      #pragma unroll
      for (int kk = 0; kk < 8; ++kk) acc = WMMA_BF16(xa[kk].v, bB[kk].v, acc);
      SGB(0x020, 16, 0);
      SGB(0x008, 8, 0);
      tileStore(acc, nt + 1);
    }
  }
  __syncthreads();   // K/V visible to all waves

  // ---- Phase C: per-head attention + fused output projection ----
  const float scale = 0.17677669529663687f;  // 1/sqrt(32)
  v8f yacc[16];
  #pragma unroll
  for (int i = 0; i < 16; ++i) { v8f z = {}; yacc[i] = z; }

  for (int h = 0; h < 8; ++h) {
    // S = Q_h @ K_h^T : 16 ds_load_b128 clause, then 8 WMMAs
    ABf bk[8];
    #pragma unroll
    for (int nt = 0; nt < 8; ++nt) {
      const unsigned short* kp = sK + (nt * 16 + l16) * 264 + h * 32 + half * 8;
      bk[nt].h[0] = *(const v8us*)(kp);
      bk[nt].h[1] = *(const v8us*)(kp + 16);
    }
    v8f s[8];
    #pragma unroll
    for (int nt = 0; nt < 8; ++nt) {
      v8f z = {};
      s[nt] = WMMA_BF16(qa[h].v, bk[nt].v, z);
    }
    SGB(0x100, 16, 0);   // 16 DS reads
    SGB(0x008, 8, 0);    // then 8 WMMAs

    // softmax over 128 cols: per-row reduce (8 tiles in regs + 16-lane butterfly)
    float rinv[8];
    #pragma unroll
    for (int i = 0; i < 8; ++i) {
      float m = s[0][i];
      #pragma unroll
      for (int nt = 1; nt < 8; ++nt) m = fmaxf(m, s[nt][i]);
      m = fmaxf(m, __shfl_xor(m, 1, 32));
      m = fmaxf(m, __shfl_xor(m, 2, 32));
      m = fmaxf(m, __shfl_xor(m, 4, 32));
      m = fmaxf(m, __shfl_xor(m, 8, 32));
      float sum = 0.f;
      #pragma unroll
      for (int nt = 0; nt < 8; ++nt) {
        float e = __expf((s[nt][i] - m) * scale);
        s[nt][i] = e;
        sum += e;
      }
      sum += __shfl_xor(sum, 1, 32);
      sum += __shfl_xor(sum, 2, 32);
      sum += __shfl_xor(sum, 4, 32);
      sum += __shfl_xor(sum, 8, 32);
      rinv[i] = 1.0f / sum;
    }

    // O = P @ V_h : hoist all V fragments (independent of scratch), then per-chunk
    // P-transpose round-trip + 2 WMMAs.
    ABf bv[8];
    #pragma unroll
    for (int kc = 0; kc < 4; ++kc) {
      const unsigned short* vp0 = sV + (h * 32 + l16) * 136 + kc * 32 + half * 8;
      const unsigned short* vp1 = sV + (h * 32 + 16 + l16) * 136 + kc * 32 + half * 8;
      bv[kc * 2].h[0]     = *(const v8us*)(vp0);
      bv[kc * 2].h[1]     = *(const v8us*)(vp0 + 16);
      bv[kc * 2 + 1].h[0] = *(const v8us*)(vp1);
      bv[kc * 2 + 1].h[1] = *(const v8us*)(vp1 + 16);
    }
    v8f o0 = {}, o1 = {};
    #pragma unroll
    for (int kc = 0; kc < 4; ++kc) {
      #pragma unroll
      for (int ht = 0; ht < 2; ++ht) {
        int nt = kc * 2 + ht;
        #pragma unroll
        for (int i = 0; i < 8; ++i)
          scr[(i + half * 8) * 32 + ht * 16 + l16] = f2bf(s[nt][i]);
      }
      asm volatile("s_wait_dscnt 0x0" ::: "memory");
      ABf ap;
      ap.h[0] = *(const v8us*)(scr + l16 * 32 + half * 8);
      ap.h[1] = *(const v8us*)(scr + l16 * 32 + 16 + half * 8);
      o0 = WMMA_BF16(ap.v, bv[kc * 2].v, o0);
      o1 = WMMA_BF16(ap.v, bv[kc * 2 + 1].v, o1);
    }
    #pragma unroll
    for (int i = 0; i < 8; ++i) { o0[i] *= rinv[i]; o1[i] *= rinv[i]; }

    // O_h -> A-layout; Y += O_h @ WoutT rows (K-slice = h*32..h*32+31)
    #pragma unroll
    for (int i = 0; i < 8; ++i) {
      scr[(i + half * 8) * 32 + l16]      = f2bf(o0[i]);
      scr[(i + half * 8) * 32 + 16 + l16] = f2bf(o1[i]);
    }
    asm volatile("s_wait_dscnt 0x0" ::: "memory");
    ABf oa;
    oa.h[0] = *(const v8us*)(scr + l16 * 32 + half * 8);
    oa.h[1] = *(const v8us*)(scr + l16 * 32 + 16 + half * 8);

    const unsigned short* wp = woutT + h * 32 + half * 8;
    #pragma unroll
    for (int g = 0; g < 2; ++g) {
      ABf bw[8];
      #pragma unroll
      for (int t = 0; t < 8; ++t) {
        const unsigned short* q = wp + (size_t)((g * 8 + t) * 16 + l16) * 256;
        bw[t].h[0] = *(const v8us*)(q);
        bw[t].h[1] = *(const v8us*)(q + 16);
      }
      #pragma unroll
      for (int t = 0; t < 8; ++t)
        yacc[g * 8 + t] = WMMA_BF16(oa.v, bw[t].v, yacc[g * 8 + t]);
      SGB(0x020, 16, 0);   // 16 VMEM reads
      SGB(0x008, 8, 0);    // then 8 WMMAs
    }
  }

  // ---- Phase D: epilogue: bias + write (pass0) or accumulate (pass1) ----
  float* orow = out + base;
  #pragma unroll
  for (int nt = 0; nt < 16; ++nt) {
    int col = nt * 16 + l16;
    float bc = bout[col];
    #pragma unroll
    for (int i = 0; i < 8; ++i) {
      int row = r0 + i + half * 8;
      size_t idx = (size_t)row * tokStride + col;
      float v = yacc[nt][i] + bc;
      if (mode) orow[idx] += v;
      else      orow[idx]  = v;
    }
  }
}

extern "C" void kernel_launch(void* const* d_in, const int* in_sizes, int n_in,
                              void* d_out, int out_size, void* d_ws, size_t ws_size,
                              hipStream_t stream) {
  const float* x     = (const float*)d_in[0];
  const float* Wq0   = (const float*)d_in[1];
  const float* Wkv0  = (const float*)d_in[2];
  const float* Wout0 = (const float*)d_in[3];
  const float* bout0 = (const float*)d_in[4];
  const float* Wq1   = (const float*)d_in[5];
  const float* Wkv1  = (const float*)d_in[6];
  const float* Wout1 = (const float*)d_in[7];
  const float* bout1 = (const float*)d_in[8];
  float* out = (float*)d_out;

  unsigned short* wbf = (unsigned short*)d_ws;
  const int QKV = 768 * 256, OUT = 256 * 256;
  const int total = 2 * QKV + 2 * OUT;

  prep_weights_kernel<<<(total + 255) / 256, 256, 0, stream>>>(
      Wq0, Wkv0, Wout0, Wq1, Wkv1, Wout1, wbf);

  // pass 0: attention along H (writes out)
  axial_attn_kernel<<<1024, 256, 0, stream>>>(
      x, wbf, wbf + 2 * QKV, bout0, out, 0);
  // pass 1: attention along W (accumulates into out)
  axial_attn_kernel<<<1024, 256, 0, stream>>>(
      x, wbf + QKV, wbf + 2 * QKV + OUT, bout1, out, 1);
}
